// OmniSoftMax_12704513261777
// MI455X (gfx1250) — compile-verified, matrix operand order checked by reading, not run on hardware
//
#include <hip/hip_runtime.h>
#include <hip/hip_bf16.h>
#include <math.h>

#define NUM_CLASSES 100000
#define NUM_FEATURES 256
#define BATCH 1024
#define MOMENTUM 0.5f
#define SCALAR 30.0f
#define OIM_EPS 1e-12f

typedef __bf16 bf16_t;
typedef __attribute__((ext_vector_type(8)))  __bf16 v8bf;
typedef __attribute__((ext_vector_type(16))) __bf16 v16bf;
typedef __attribute__((ext_vector_type(8)))  float  v8f;
typedef __attribute__((ext_vector_type(4)))  float  v4f;

union V16U { v16bf v; v8bf h[2]; };

// ---------------------------------------------------------------------------
// Normalize inputs: one block (256 threads) per row. Writes f32 (for the
// momentum-update kernel) and bf16 (WMMA A operand).
// ---------------------------------------------------------------------------
__global__ void oim_norm_x(const float* __restrict__ x,
                           float* __restrict__ xn,
                           bf16_t* __restrict__ xnb) {
    __shared__ float red[256];
    const int row = blockIdx.x, t = threadIdx.x;
    const float v = x[row * NUM_FEATURES + t];
    red[t] = v * v;
    __syncthreads();
    for (int s = 128; s > 0; s >>= 1) {
        if (t < s) red[t] += red[t + s];
        __syncthreads();
    }
    const float inv = 1.0f / fmaxf(sqrtf(red[0]), OIM_EPS);
    const float nv = v * inv;
    xn[row * NUM_FEATURES + t] = nv;
    xnb[row * NUM_FEATURES + t] = (bf16_t)nv;
}

// ---------------------------------------------------------------------------
// Normalize weight rows -> bf16 (WMMA B operand), staged in the new_weight
// output region (overwritten after the GEMM by the raw copy).
// ---------------------------------------------------------------------------
__global__ void oim_norm_w(const float* __restrict__ w,
                           bf16_t* __restrict__ wnb) {
    __shared__ float red[256];
    const int row = blockIdx.x, t = threadIdx.x;
    const float v = w[(size_t)row * NUM_FEATURES + t];
    red[t] = v * v;
    __syncthreads();
    for (int s = 128; s > 0; s >>= 1) {
        if (t < s) red[t] += red[t + s];
        __syncthreads();
    }
    const float inv = 1.0f / fmaxf(sqrtf(red[0]), OIM_EPS);
    wnb[(size_t)row * NUM_FEATURES + t] = (bf16_t)(v * inv);
}

// ---------------------------------------------------------------------------
// predicts = (x̂ @ ŵᵀ) * 30 via v_wmma_f32_16x16x32_bf16.
// Block = 256 threads = 8 waves. Wave w owns M rows [mbase, mbase+32) as two
// 16-row tiles, and N columns [nbase, nbase+64) as four 16-col tiles.
// K loop: 256 / 32 = 8 steps; A fragments reused across the 4 B tiles.
// A/B per-lane packing per ISA §7.12.2 (16-bit 16x32 operand layout).
// ---------------------------------------------------------------------------
__global__ void __launch_bounds__(256)
oim_gemm(const bf16_t* __restrict__ A,   // [1024 x 256] normalized x, bf16
         const bf16_t* __restrict__ Bm,  // [100000 x 256] normalized w, bf16
         float* __restrict__ out) {      // [1024 x 100000]
    const int lane = threadIdx.x & 31;
    const int w    = threadIdx.x >> 5;
    const int r    = lane & 15;
    const int hi   = lane >> 4;

    const int nbase = blockIdx.x * 64;
    const int mbase = blockIdx.y * 256 + w * 32;

    const v8f vzero = {0.f, 0.f, 0.f, 0.f, 0.f, 0.f, 0.f, 0.f};
    v8f acc[2][4];
#pragma unroll
    for (int m = 0; m < 2; ++m)
#pragma unroll
        for (int t = 0; t < 4; ++t) acc[m][t] = vzero;

    const bf16_t* arow0 = A + (size_t)(mbase + r) * NUM_FEATURES;
    const bf16_t* arow1 = A + (size_t)(mbase + 16 + r) * NUM_FEATURES;
    const bf16_t* brow[4];
#pragma unroll
    for (int t = 0; t < 4; ++t) {
        int n = nbase + t * 16 + r;
        if (n >= NUM_CLASSES) n = NUM_CLASSES - 1;  // clamp (store is guarded)
        brow[t] = Bm + (size_t)n * NUM_FEATURES;
    }

#pragma unroll
    for (int kb = 0; kb < NUM_FEATURES; kb += 32) {
        V16U a0, a1;
        a0.h[0] = *(const v8bf*)(arow0 + kb + hi * 8);
        a0.h[1] = *(const v8bf*)(arow0 + kb + 16 + hi * 8);
        a1.h[0] = *(const v8bf*)(arow1 + kb + hi * 8);
        a1.h[1] = *(const v8bf*)(arow1 + kb + 16 + hi * 8);
#pragma unroll
        for (int t = 0; t < 4; ++t) {
            V16U bt;
            bt.h[0] = *(const v8bf*)(brow[t] + kb + hi * 8);
            bt.h[1] = *(const v8bf*)(brow[t] + kb + 16 + hi * 8);
            acc[0][t] = __builtin_amdgcn_wmma_f32_16x16x32_bf16(
                false, a0.v, false, bt.v, (short)0, acc[0][t], false, false);
            acc[1][t] = __builtin_amdgcn_wmma_f32_16x16x32_bf16(
                false, a1.v, false, bt.v, (short)0, acc[1][t], false, false);
        }
    }

    // C/D layout (ISA §7.12.2): lanes 0-15 -> M = i, N = lane;
    // lanes 16-31 -> M = 8+i, N = lane-16.  Streaming (non-temporal) stores.
#pragma unroll
    for (int t = 0; t < 4; ++t) {
        const int n = nbase + t * 16 + r;
        if (n < NUM_CLASSES) {
#pragma unroll
            for (int m = 0; m < 2; ++m) {
                const size_t rowb = (size_t)(mbase + m * 16 + hi * 8) * NUM_CLASSES + n;
#pragma unroll
                for (int i = 0; i < 8; ++i) {
                    __builtin_nontemporal_store(acc[m][t][i] * SCALAR,
                                                out + rowb + (size_t)i * NUM_CLASSES);
                }
            }
        }
    }
}

// ---------------------------------------------------------------------------
// Raw weight -> new_weight output region (16B streaming copy, native vector
// type so __builtin_nontemporal_store accepts it).
// ---------------------------------------------------------------------------
__global__ void oim_copy_w(const v4f* __restrict__ src, v4f* __restrict__ dst) {
    const size_t i = (size_t)blockIdx.x * blockDim.x + threadIdx.x;
    if (i < (size_t)NUM_CLASSES * NUM_FEATURES / 4) {
        const v4f v = src[i];
        __builtin_nontemporal_store(v, &dst[i]);
    }
}

__global__ void oim_copy_tgt(const int* __restrict__ t, int* __restrict__ o) {
    const int i = blockIdx.x * blockDim.x + threadIdx.x;
    if (i < BATCH) o[i] = t[i];
}

// ---------------------------------------------------------------------------
// Sequential momentum scan, parallelized by class chain: block b proceeds only
// if sample b is the FIRST occurrence of its class, then replays all samples
// of that class in order (row held in registers across the chain).
// ---------------------------------------------------------------------------
__global__ void oim_update(const int* __restrict__ tgt,
                           const float* __restrict__ xn,
                           float* __restrict__ W) {
    __shared__ float red[256];
    const int b = blockIdx.x, t = threadIdx.x;
    const int y = tgt[b];
    for (int j = 0; j < b; ++j)
        if (tgt[j] == y) return;  // uniform exit: not the chain owner

    float wv = W[(size_t)y * NUM_FEATURES + t];
    for (int j = b; j < BATCH; ++j) {
        if (tgt[j] != y) continue;  // uniform branch
        const float nv = MOMENTUM * wv + (1.0f - MOMENTUM) * xn[j * NUM_FEATURES + t];
        red[t] = nv * nv;
        __syncthreads();
        for (int s = 128; s > 0; s >>= 1) {
            if (t < s) red[t] += red[t + s];
            __syncthreads();
        }
        wv = nv / fmaxf(sqrtf(red[0]), OIM_EPS);
        __syncthreads();  // red[] reused next chain step
    }
    W[(size_t)y * NUM_FEATURES + t] = wv;
}

// ---------------------------------------------------------------------------
extern "C" void kernel_launch(void* const* d_in, const int* in_sizes, int n_in,
                              void* d_out, int out_size, void* d_ws, size_t ws_size,
                              hipStream_t stream) {
    const float* x   = (const float*)d_in[0];
    const int*   tgt = (const int*)d_in[1];
    const float* wgt = (const float*)d_in[2];

    float* out      = (float*)d_out;
    float* predicts = out;                                              // 1024*100000
    int*   out_tgt  = (int*)(out + (size_t)BATCH * NUM_CLASSES);        // 1024
    float* out_w    = out + (size_t)BATCH * NUM_CLASSES + BATCH;        // 100000*256

    float*  xn  = (float*)d_ws;                                         // 1 MB
    bf16_t* xnb = (bf16_t*)((char*)d_ws + (size_t)BATCH * NUM_FEATURES * sizeof(float));
    bf16_t* wnb = (bf16_t*)out_w;  // stage bf16 ŵ in new_weight region (pre-GEMM)

    oim_norm_x<<<BATCH, 256, 0, stream>>>(x, xn, xnb);
    oim_norm_w<<<NUM_CLASSES, 256, 0, stream>>>(wgt, wnb);

    dim3 grid((NUM_CLASSES + 63) / 64, BATCH / 256);  // (1563, 4)
    oim_gemm<<<grid, 256, 0, stream>>>(xnb, wnb, predicts);

    const int nvec4 = NUM_CLASSES * NUM_FEATURES / 4;  // 6.4M float4
    oim_copy_w<<<(nvec4 + 255) / 256, 256, 0, stream>>>((const v4f*)wgt, (v4f*)out_w);
    oim_copy_tgt<<<(BATCH + 255) / 256, 256, 0, stream>>>(tgt, out_tgt);
    oim_update<<<BATCH, 256, 0, stream>>>(tgt, xn, out_w);
}
